// SnakeLoss_51316269253468
// MI455X (gfx1250) — compile-verified, matrix operand order checked by reading, not run on hardware
//
#include <hip/hip_runtime.h>
#include <hip/hip_bf16.h>

typedef __attribute__((ext_vector_type(16))) _Float16 v16h;
typedef __attribute__((ext_vector_type(8)))  _Float16 v8h;
typedef __attribute__((ext_vector_type(8)))  float    v8f;

// B=32, N=1024, D=2 -> per-batch flattened length 2048.
// One workgroup (128 threads = 4 wave32) per batch.
// Wave w computes corr for shifts [256w, 256w+256) via block-Toeplitz WMMA.
__global__ __launch_bounds__(128)
void snake_corr_kernel(const float* __restrict__ input,
                       const float* __restrict__ target,
                       float* __restrict__ batch_loss) {
    const int b   = blockIdx.x;
    const int tid = threadIdx.x;

    // x extended: indices m0+16i+k+16 <= 992+240+16+8+7 = 1263  -> 1312 (16B-aligned rows)
    // t extended (Toeplitz, offset by +1024): accessed range [241, 2047] -> 2048
    __shared__ _Float16 xh[2][1312] __attribute__((aligned(16)));
    __shared__ _Float16 th[2][2048] __attribute__((aligned(16)));
    __shared__ float red[128];
    __shared__ float s_total;

    const float* __restrict__ xg = input  + b * 2048;
    const float* __restrict__ tg = target + b * 2048;

    // ---- Stage to LDS as f16 with circular extension (layout (n,d) -> [d][n]) ----
    for (int p = tid; p < 2 * 1312; p += 128) {
        int d = p / 1312, q = p - d * 1312;
        xh[d][q] = (_Float16)xg[((q & 1023) << 1) + d];
    }
    for (int p = tid; p < 2 * 2048; p += 128) {
        int d = p >> 11, q = p & 2047;
        th[d][q] = (_Float16)tg[((q & 1023) << 1) + d];   // th[d][q] = t[(q-1024) mod N]
    }

    // ---- ||x||^2 + ||t||^2 in exact f32 ----
    float ss = 0.0f;
    for (int i = tid; i < 2048; i += 128) {
        float a = xg[i], c = tg[i];
        ss = fmaf(a, a, ss);
        ss = fmaf(c, c, ss);
    }
    red[tid] = ss;
    __syncthreads();
    for (int s = 64; s > 0; s >>= 1) {
        if (tid < s) red[tid] += red[tid + s];
        __syncthreads();
    }
    if (tid == 0) s_total = red[0];
    __syncthreads();

    // ---- WMMA phase: D[i,j] = sum_k x[m0+16i+k] * t[m0+k-s0-j]  -> corr[s0+16i+j] ----
    const int lane  = tid & 31;
    const int wave  = tid >> 5;
    const int row   = lane & 15;            // A: row i, B: column j
    const int off8  = ((lane >> 4) & 1) << 3; // K-half select per documented 16-bit layout
    const int s0    = wave << 8;            // 256 shifts per wave
    const int tbase = 1024 + off8 - s0 - row;

    v8f acc = {};
    for (int d = 0; d < 2; ++d) {
        const _Float16* __restrict__ xp = &xh[d][0];
        const _Float16* __restrict__ tp = &th[d][0];
#pragma unroll 4
        for (int m0 = 0; m0 < 1024; m0 += 32) {
            // A tile: two aligned 16B runs per lane (K = off8..off8+7, 16+off8..16+off8+7)
            const _Float16* xa = xp + m0 + (row << 4) + off8;
            v8h lo = *(const v8h*)(xa);
            v8h hi = *(const v8h*)(xa + 16);
            v16h A;
#pragma unroll
            for (int p = 0; p < 8; ++p) { A[p] = lo[p]; A[p + 8] = hi[p]; }

            // B tile: Toeplitz gather (unaligned by j) -> u16 LDS loads
            const _Float16* tb = tp + (tbase + m0);
            v16h Bm;
#pragma unroll
            for (int p = 0; p < 8; ++p) { Bm[p] = tb[p]; Bm[p + 8] = tb[16 + p]; }

            acc = __builtin_amdgcn_wmma_f32_16x16x32_f16(
                false, A, false, Bm, (short)0, acc, false, false);
        }
    }

    // ---- max over all shifts: per-thread max of 8 accumulators, then block reduce ----
    float m = acc[0];
#pragma unroll
    for (int v = 1; v < 8; ++v) m = fmaxf(m, acc[v]);
    red[tid] = m;
    __syncthreads();
    for (int s = 64; s > 0; s >>= 1) {
        if (tid < s) red[tid] = fmaxf(red[tid], red[tid + s]);
        __syncthreads();
    }
    if (tid == 0)
        batch_loss[b] = (s_total - 2.0f * red[0]) * (1.0f / 2048.0f);
}

__global__ __launch_bounds__(32)
void snake_mean_kernel(const float* __restrict__ batch_loss, float* __restrict__ out) {
    __shared__ float r[32];
    int t = threadIdx.x;
    r[t] = batch_loss[t];
    __syncthreads();
    for (int s = 16; s > 0; s >>= 1) {
        if (t < s) r[t] += r[t + s];
        __syncthreads();
    }
    if (t == 0) out[0] = r[0] * (1.0f / 32.0f);
}

extern "C" void kernel_launch(void* const* d_in, const int* in_sizes, int n_in,
                              void* d_out, int out_size, void* d_ws, size_t ws_size,
                              hipStream_t stream) {
    const float* input  = (const float*)d_in[0];
    const float* target = (const float*)d_in[1];
    float* out = (float*)d_out;
    float* bl  = (float*)d_ws;   // 32 per-batch losses

    hipLaunchKernelGGL(snake_corr_kernel, dim3(32), dim3(128), 0, stream,
                       input, target, bl);
    hipLaunchKernelGGL(snake_mean_kernel, dim3(1), dim3(32), 0, stream,
                       bl, out);
}
